// Net_atomic_41300405518869
// MI455X (gfx1250) — matrix operand order repaired and checked by reading, not compiled, not run on hardware
//
#include <hip/hip_runtime.h>

typedef __attribute__((ext_vector_type(16))) _Float16 v16h;
typedef __attribute__((ext_vector_type(8)))  _Float16 v8h;
typedef __attribute__((ext_vector_type(8)))  float    v8f;

#define N_ATOMS 32768
#define K_NBR   32
#define HID     64
#define HC      16
#define FDIM    64
#define GDIM    50
#define LLAYERS 3
#define NGRAPH  8
#define LOG2E_  1.44269504089f
#define LN2_    0.69314718056f

// ShiftedSoftplus with raw hardware exp2/log2 (arg of log is in [1,2],
// so no subnormal guard needed):
//   ssp(x) = max(x,0) + ln2*(log2(1 + 2^(-|x|*log2e)) - 1)
__device__ __forceinline__ float sspf(float x) {
  float e = __builtin_amdgcn_exp2f(-fabsf(x) * LOG2E_);
  return fmaxf(x, 0.f) + LN2_ * (__builtin_amdgcn_logf(1.f + e) - 1.f);
}

// exp(x) for x <= 0 (gaussian / softmax weights): single v_exp_f32
__device__ __forceinline__ float fexp(float x) {
  return __builtin_amdgcn_exp2f(x * LOG2E_);
}

__device__ __forceinline__ v16h cat16(v8h lo, v8h hi) {
  return __builtin_shufflevector(lo, hi, 0, 1, 2, 3, 4, 5, 6, 7,
                                 8, 9, 10, 11, 12, 13, 14, 15);
}

// D(16x16,f32) += A(16x64 f16, row stride 64) x Wt (64x64 f16, transposed
// Wt[n*64+k]). Two V_WMMA_F32_16X16X32_F16 issues. All fragment loads are
// 16B-aligned vector loads (ds_load_b128 / global_load_b128).
__device__ __forceinline__ v8f wmma_dot64(const _Float16* A, const _Float16* Wt,
                                          int rowBase, int colBase, v8f acc) {
  const int lane = threadIdx.x & 31;
  const int hf   = lane >> 4;
  const int l    = lane & 15;
  const _Float16* arow = A  + (rowBase + l) * 64 + hf * 8;
  const _Float16* brow = Wt + (colBase + l) * 64 + hf * 8;
#pragma unroll
  for (int kc = 0; kc < 64; kc += 32) {
    v8h alo = *(const v8h*)(arow + kc);
    v8h ahi = *(const v8h*)(arow + kc + 16);
    v8h blo = *(const v8h*)(brow + kc);
    v8h bhi = *(const v8h*)(brow + kc + 16);
    acc = __builtin_amdgcn_wmma_f32_16x16x32_f16(
        /*neg_a=*/false, cat16(alo, ahi), /*neg_b=*/false, cat16(blo, bhi),
        /*c_mod=*/(short)0, acc, /*reuse_a=*/false, /*reuse_b=*/false);
  }
  return acc;
}

// ---- weight prep: f32 [K,N] -> f16 transposed [N,Kpad] (zero-padded K) ----
__global__ __launch_bounds__(256) void prep_transpose_f16(
    const float* __restrict__ W, _Float16* __restrict__ Wt,
    int Kdim, int Ncols, int Kpad) {
  int idx = blockIdx.x * 256 + threadIdx.x;
  if (idx >= Ncols * Kpad) return;
  int n = idx / Kpad, kk = idx - n * Kpad;
  Wt[idx] = (kk < Kdim) ? (_Float16)W[kk * Ncols + n] : (_Float16)0.0f;
}

// ---- x = emb[z], plus f16 mirror ----
__global__ __launch_bounds__(256) void init_embed(
    const int* __restrict__ z, const float* __restrict__ emb,
    float* __restrict__ x, _Float16* __restrict__ xh) {
  int idx = blockIdx.x * 256 + threadIdx.x;  // N*64 total
  int n = idx >> 6, c = idx & 63;
  float vv = emb[z[n] * HID + c];
  x[idx] = vv;
  xh[idx] = (_Float16)vv;
}

// ---- node GEMM, epilogue specialized at compile time ----
template <bool RES, bool ACT, bool WF, bool WH>
__global__ __launch_bounds__(256) void node_gemm_t(
    const _Float16* __restrict__ A, const _Float16* __restrict__ Wt,
    const float* __restrict__ bias, const float* __restrict__ resid,
    float* __restrict__ outF, _Float16* __restrict__ outH, int ncols) {
  const int t = threadIdx.x;
  const int w = t >> 5, lane = t & 31, hf = lane >> 4, l = lane & 15;
  const int rowBase = blockIdx.x * 128 + w * 16;
  const int ntiles = ncols >> 4;
  for (int ct = 0; ct < ntiles; ++ct) {
    v8f acc = {};
    acc = wmma_dot64(A, Wt, rowBase, ct * 16, acc);
    const int col = ct * 16 + l;
    const float bcol = bias[col];
#pragma unroll
    for (int i = 0; i < 8; ++i) {
      int row = rowBase + hf * 8 + i;
      float vv = acc[i] + bcol;
      if (RES) vv += resid[row * ncols + col];
      if (ACT) vv = sspf(vv);
      if (WF) outF[row * ncols + col] = vv;
      if (WH) outH[row * ncols + col] = (_Float16)vv;
    }
  }
}

// ---- fused edge pipeline: 4 dst nodes (128 edges) per block ----
// dst = repeat(arange(N),K) -> each node's 32 edges are contiguous.
__global__ __launch_bounds__(256) void edge_attention(
    const float* __restrict__ pos, const int* __restrict__ srcIdx,
    const float* __restrict__ q, const float* __restrict__ k,
    const float* __restrict__ v,
    const _Float16* __restrict__ w1p, const _Float16* __restrict__ w2t,
    const _Float16* __restrict__ wet,
    const float* __restrict__ b1, const float* __restrict__ b2,
    float* __restrict__ hagg) {
  __shared__ _Float16 sA[128 * 64];  // gaussians, then e
  __shared__ _Float16 sB[128 * 64];  // h1, then eproj
  __shared__ _Float16 sW[64 * 64];   // current weight panel (transposed)
  __shared__ float    sD[128];
  __shared__ int      sS[128];
  __shared__ float    sQ[4 * 64];
  __shared__ float    sAl[128 * 4];

  const int t  = threadIdx.x;
  const int nb = blockIdx.x * 4;   // first dst node
  const int eb = nb * K_NBR;       // first edge
  const int w = t >> 5, lane = t & 31, hf = lane >> 4, l = lane & 15;
  const int rowBase = w * 16;

  // phase 0: src ids, distances, q rows for the 4 dst nodes
  if (t < 128) {
    int s = srcIdx[eb + t];
    sS[t] = s;
    int dn = nb + (t >> 5);
    float dx = pos[s * 3 + 0] - pos[dn * 3 + 0];
    float dy = pos[s * 3 + 1] - pos[dn * 3 + 1];
    float dz = pos[s * 3 + 2] - pos[dn * 3 + 2];
    sD[t] = sqrtf(dx * dx + dy * dy + dz * dz + 1e-12f);
  }
  sQ[t] = q[(nb + (t >> 6)) * HID + (t & 63)];
  __syncthreads();

  // phase 1: GaussianSmearing into sA (K padded 50 -> 64 with zeros)
  const float step  = 5.0f / (GDIM - 1);
  const float coeff = -0.5f / (step * step);
  for (int idx = t; idx < 128 * 64; idx += 256) {
    int e = idx >> 6, g = idx & 63;
    float vv = 0.0f;
    if (g < GDIM) { float dd = sD[e] - (float)g * step; vv = fexp(coeff * dd * dd); }
    sA[idx] = (_Float16)vv;
  }
  for (int idx = t; idx < 64 * 64; idx += 256) sW[idx] = w1p[idx];
  __syncthreads();

  // GEMM1: h1 = ssp(attr @ W1 + b1) -> sB
  for (int ct = 0; ct < 4; ++ct) {
    v8f acc = {};
    acc = wmma_dot64(sA, sW, rowBase, ct * 16, acc);
    const int col = ct * 16 + l;
    const float bcol = b1[col];
#pragma unroll
    for (int i = 0; i < 8; ++i) {
      int row = rowBase + hf * 8 + i;
      sB[row * 64 + col] = (_Float16)sspf(acc[i] + bcol);
    }
  }
  __syncthreads();
  for (int idx = t; idx < 64 * 64; idx += 256) sW[idx] = w2t[idx];
  __syncthreads();

  // GEMM2: e = h1 @ W2 + b2 -> sA
  for (int ct = 0; ct < 4; ++ct) {
    v8f acc = {};
    acc = wmma_dot64(sB, sW, rowBase, ct * 16, acc);
    const int col = ct * 16 + l;
    const float bcol = b2[col];
#pragma unroll
    for (int i = 0; i < 8; ++i) {
      int row = rowBase + hf * 8 + i;
      sA[row * 64 + col] = (_Float16)(acc[i] + bcol);
    }
  }
  __syncthreads();
  for (int idx = t; idx < 64 * 64; idx += 256) sW[idx] = wet[idx];
  __syncthreads();

  // GEMM3: eproj = e @ We -> sB
  for (int ct = 0; ct < 4; ++ct) {
    v8f acc = {};
    acc = wmma_dot64(sA, sW, rowBase, ct * 16, acc);
    const int col = ct * 16 + l;
#pragma unroll
    for (int i = 0; i < 8; ++i) {
      int row = rowBase + hf * 8 + i;
      sB[row * 64 + col] = (_Float16)acc[i];
    }
  }
  __syncthreads();

  // phase 2: attention logits  alpha[e,h] = 0.25 * q[dst] . (k[src]+eproj)
  {
    int et = t & 127;
    int sI = sS[et];
    int di = et >> 5;
    __builtin_prefetch(&v[sI * HID], 0, 1);
    int hb = (t >> 7) * 2;
    const float* kr = k + sI * HID;
    const float* qr = sQ + di * HID;
    for (int h = hb; h < hb + 2; ++h) {
      float dot = 0.f;
#pragma unroll
      for (int j = 0; j < HC; ++j) {
        int c = h * HC + j;
        float kk = kr[c] + (float)sB[et * 64 + c];
        dot += qr[c] * kk;
      }
      sAl[et * 4 + h] = dot * 0.25f;  // 1/sqrt(HC)
    }
  }
  __syncthreads();

  // phase 3: softmax over each dst node's 32 contiguous edges, per head
  if (t < 16) {
    int di = t >> 2, h = t & 3;
    float mx = -1e30f;
    for (int e2 = 0; e2 < 32; ++e2)
      mx = fmaxf(mx, sAl[(di * 32 + e2) * 4 + h]);
    float sum = 0.f;
    for (int e2 = 0; e2 < 32; ++e2) {
      float ex = fexp(sAl[(di * 32 + e2) * 4 + h] - mx);
      sAl[(di * 32 + e2) * 4 + h] = ex;
      sum += ex;
    }
    float inv = 1.f / (sum + 1e-16f);
    for (int e2 = 0; e2 < 32; ++e2) sAl[(di * 32 + e2) * 4 + h] *= inv;
  }
  __syncthreads();

  // phase 4: aggregate  hagg[dst,c] = sum_e a[e,h(c)] * (v[src]+eproj)[c]
  {
    int di = t >> 6, c = t & 63, h = c >> 4;
    float acc = 0.f;
    for (int e2 = 0; e2 < 32; ++e2) {
      int ee = di * 32 + e2;
      float vv = v[sS[ee] * HID + c] + (float)sB[ee * 64 + c];
      acc += vv * sAl[ee * 4 + h];
    }
    hagg[(nb + di) * HID + c] = acc;
  }
}

// ---- y[n] = h[n,:32] . out_w2 + out_b2 ----
__global__ __launch_bounds__(256) void node_y(
    const float* __restrict__ h, const float* __restrict__ w2,
    const float* __restrict__ b2, float* __restrict__ y) {
  int n = blockIdx.x * 256 + threadIdx.x;
  float acc = b2[0];
#pragma unroll
  for (int j = 0; j < 32; ++j) acc += h[n * 32 + j] * w2[j];
  y[n] = acc;
}

// ---- deterministic per-graph reduction: one block per graph ----
__global__ __launch_bounds__(256) void graph_reduce(
    const float* __restrict__ y, const int* __restrict__ batch,
    float* __restrict__ out) {
  __shared__ float sm[256];
  int g = blockIdx.x;
  float acc = 0.f;
  for (int n = threadIdx.x; n < N_ATOMS; n += 256)
    if (batch[n] == g) acc += y[n];
  sm[threadIdx.x] = acc;
  __syncthreads();
  for (int s = 128; s > 0; s >>= 1) {
    if (threadIdx.x < s) sm[threadIdx.x] += sm[threadIdx.x + s];
    __syncthreads();
  }
  if (threadIdx.x == 0) out[g] = sm[0];
}

extern "C" void kernel_launch(void* const* d_in, const int* in_sizes, int n_in,
                              void* d_out, int out_size, void* d_ws, size_t ws_size,
                              hipStream_t stream) {
  (void)in_sizes; (void)n_in; (void)out_size; (void)ws_size;
  const int*   z    = (const int*)  d_in[0];
  const float* pos  = (const float*)d_in[1];
  const int*   eidx = (const int*)  d_in[2];
  const int*   batch= (const int*)  d_in[3];
  const float* emb  = (const float*)d_in[4];
  const float* mw1  = (const float*)d_in[5];
  const float* mb1  = (const float*)d_in[6];
  const float* mw2  = (const float*)d_in[7];
  const float* mb2  = (const float*)d_in[8];
  const float* wq   = (const float*)d_in[9];
  const float* bq   = (const float*)d_in[10];
  const float* wk_  = (const float*)d_in[11];
  const float* bk   = (const float*)d_in[12];
  const float* wv   = (const float*)d_in[13];
  const float* bv   = (const float*)d_in[14];
  const float* we   = (const float*)d_in[15];
  const float* wsk  = (const float*)d_in[16];
  const float* bs   = (const float*)d_in[17];
  const float* lw   = (const float*)d_in[18];
  const float* lb   = (const float*)d_in[19];
  const float* ow1  = (const float*)d_in[20];
  const float* ob1  = (const float*)d_in[21];
  const float* ow2  = (const float*)d_in[22];
  const float* ob2  = (const float*)d_in[23];
  float* out = (float*)d_out;

  // workspace carve-out (~53 MB)
  char* base = (char*)d_ws;
  size_t off = 0;
  auto carve = [&](size_t bytes) -> void* {
    void* p = base + off;
    off = (off + bytes + 255) & ~(size_t)255;
    return p;
  };
  _Float16* xh   = (_Float16*)carve((size_t)N_ATOMS * HID * 2);
  _Float16* th   = (_Float16*)carve((size_t)N_ATOMS * HID * 2);
  float*    x    = (float*)   carve((size_t)N_ATOMS * HID * 4);
  float*    qb   = (float*)   carve((size_t)N_ATOMS * HID * 4);
  float*    kb   = (float*)   carve((size_t)N_ATOMS * HID * 4);
  float*    vb   = (float*)   carve((size_t)N_ATOMS * HID * 4);
  float*    hagg = (float*)   carve((size_t)N_ATOMS * HID * 4);
  float*    hout = (float*)   carve((size_t)N_ATOMS * 32 * 4);
  float*    yv   = (float*)   carve((size_t)N_ATOMS * 4);
  _Float16* wt   = (_Float16*)carve((size_t)(LLAYERS * 8 * 4096 + 32 * 64) * 2);

  auto wslot = [&](int layer, int m) { return wt + (size_t)(layer * 8 + m) * 4096; };
  _Float16* ow1t = wt + (size_t)LLAYERS * 8 * 4096;

  dim3 b256(256);
  for (int lyr = 0; lyr < LLAYERS; ++lyr) {
    prep_transpose_f16<<<16, b256, 0, stream>>>(mw1 + lyr * GDIM * FDIM, wslot(lyr, 0), GDIM, 64, 64);
    prep_transpose_f16<<<16, b256, 0, stream>>>(mw2 + lyr * 4096, wslot(lyr, 1), 64, 64, 64);
    prep_transpose_f16<<<16, b256, 0, stream>>>(wq  + lyr * 4096, wslot(lyr, 2), 64, 64, 64);
    prep_transpose_f16<<<16, b256, 0, stream>>>(wk_ + lyr * 4096, wslot(lyr, 3), 64, 64, 64);
    prep_transpose_f16<<<16, b256, 0, stream>>>(wv  + lyr * 4096, wslot(lyr, 4), 64, 64, 64);
    prep_transpose_f16<<<16, b256, 0, stream>>>(we  + lyr * 4096, wslot(lyr, 5), 64, 64, 64);
    prep_transpose_f16<<<16, b256, 0, stream>>>(wsk + lyr * 4096, wslot(lyr, 6), 64, 64, 64);
    prep_transpose_f16<<<16, b256, 0, stream>>>(lw  + lyr * 4096, wslot(lyr, 7), 64, 64, 64);
  }
  prep_transpose_f16<<<8, b256, 0, stream>>>(ow1, ow1t, 64, 32, 64);

  init_embed<<<N_ATOMS * HID / 256, b256, 0, stream>>>(z, emb, x, xh);

  const int* srcIdx = eidx;  // edge_index row 0
  for (int lyr = 0; lyr < LLAYERS; ++lyr) {
    // q/k/v projections: bias, no resid, no act, f32 out
    node_gemm_t<false, false, true, false><<<N_ATOMS / 128, b256, 0, stream>>>(
        xh, wslot(lyr, 2), bq + lyr * HID, nullptr, qb, nullptr, 64);
    node_gemm_t<false, false, true, false><<<N_ATOMS / 128, b256, 0, stream>>>(
        xh, wslot(lyr, 3), bk + lyr * HID, nullptr, kb, nullptr, 64);
    node_gemm_t<false, false, true, false><<<N_ATOMS / 128, b256, 0, stream>>>(
        xh, wslot(lyr, 4), bv + lyr * HID, nullptr, vb, nullptr, 64);
    edge_attention<<<N_ATOMS / 4, b256, 0, stream>>>(
        pos, srcIdx, qb, kb, vb, wslot(lyr, 0), wslot(lyr, 1), wslot(lyr, 5),
        mb1 + lyr * FDIM, mb2 + lyr * FDIM, hagg);
    // t = ssp(hagg + x@ws + bs): bias, resid, act, f16 out
    node_gemm_t<true, true, false, true><<<N_ATOMS / 128, b256, 0, stream>>>(
        xh, wslot(lyr, 6), bs + lyr * HID, hagg, nullptr, th, 64);
    // x = x + t@lin_w + lin_b: bias, resid(x), no act, f32+f16 out
    node_gemm_t<true, false, true, true><<<N_ATOMS / 128, b256, 0, stream>>>(
        th, wslot(lyr, 7), lb + lyr * HID, x, x, xh, 64);
  }
  // readout: h = ssp(x @ out_w1 + out_b1)  [N,32]
  node_gemm_t<false, true, true, false><<<N_ATOMS / 128, b256, 0, stream>>>(
      xh, ow1t, ob1, nullptr, hout, nullptr, 32);
  node_y<<<N_ATOMS / 256, b256, 0, stream>>>(hout, ow2, ob2, yv);
  graph_reduce<<<NGRAPH, b256, 0, stream>>>(yv, batch, out);
}
